// UGRNNCell_824633720896
// MI455X (gfx1250) — matrix operand order repaired and checked by reading, not compiled, not run on hardware
//
#include <hip/hip_runtime.h>

#define B_    64
#define T_    512
#define U_    1024
#define TWOU_ 2048

typedef __attribute__((ext_vector_type(16))) __bf16 v16bf;
typedef __attribute__((ext_vector_type(8)))  __bf16 v8bf;
typedef __attribute__((ext_vector_type(4)))  __bf16 v4bf;
typedef __attribute__((ext_vector_type(8)))  float  v8f;
typedef __attribute__((ext_vector_type(4)))  float  v4f;

// A-fragment (16x32 bf16, row = lane%16): elements 0..15 = K kb..kb+15,
// where kb = k + 16*(lane/16).
__device__ __forceinline__ v16bf load_a_bf16(const __bf16* p) {
  return *(const v16bf*)(p);
}

__device__ __forceinline__ v8f wmma_bf16(v16bf a, v16bf b, v8f c) {
  return __builtin_amdgcn_wmma_f32_16x16x32_bf16(
      /*neg_a=*/false, a, /*neg_b=*/false, b,
      /*c_mod=*/(short)0, c, /*reuse_a=*/false, /*reuse_b=*/false);
}

// ---------------------------------------------------------------------------
// Kernel 1: W fp32 [2048][2048] -> WT bf16 [n][k] (transposed), LDS-tiled.
// ---------------------------------------------------------------------------
__global__ __launch_bounds__(256) void k_wconv(const float* __restrict__ W,
                                               unsigned short* __restrict__ WTo) {
  __shared__ float tile[32][33];
  __bf16* WT = (__bf16*)WTo;
  const int n0 = blockIdx.x * 32;
  const int k0 = blockIdx.y * 32;
  const int tx = threadIdx.x, ty = threadIdx.y;  // 32 x 8
#pragma unroll
  for (int i = 0; i < 32; i += 8)
    tile[ty + i][tx] = W[(size_t)(k0 + ty + i) * TWOU_ + (n0 + tx)];
  __syncthreads();
#pragma unroll
  for (int i = 0; i < 32; i += 8)
    WT[(size_t)(n0 + ty + i) * TWOU_ + (k0 + tx)] = (__bf16)tile[tx][ty + i];
}

// ---------------------------------------------------------------------------
// Kernel 2: x fp32 -> bf16 (the 64 MB bf16 copy lives in the 192 MB L2).
// ---------------------------------------------------------------------------
__global__ __launch_bounds__(256) void k_xconv(const float* __restrict__ x,
                                               unsigned short* __restrict__ Xo) {
  size_t i = ((size_t)blockIdx.x * 256 + threadIdx.x) * 4;
  v4f f = *(const v4f*)(x + i);
  v4bf b;
#pragma unroll
  for (int j = 0; j < 4; ++j) b[j] = (__bf16)f[j];
  *(v4bf*)((__bf16*)Xo + i) = b;
}

// ---------------------------------------------------------------------------
// Kernel 3: h0 -> fp32 state buffer + bf16 A-operand buffer.
// ---------------------------------------------------------------------------
__global__ __launch_bounds__(256) void k_hinit(const float* __restrict__ h0,
                                               float* __restrict__ hf,
                                               unsigned short* __restrict__ hb) {
  int i = blockIdx.x * 256 + threadIdx.x;  // 65536 elements
  float v = h0[i];
  hf[i] = v;
  ((__bf16*)hb)[i] = (__bf16)v;
}

// ---------------------------------------------------------------------------
// Kernel 4: Mx[B*T][2048] = Xbf16[B*T][1024] @ W_x + bias  (time-independent)
// 8 waves/WG arranged 4(M) x 2(N); 128x128 tile per WG; 32x64 per wave
// (2 A-frags x 4 B-frags -> 8 accumulators; 12 b128 loads per 8 WMMAs).
// Mx is write-once/read-once (256 MB > L2): stream it with NT stores so the
// Xb+WT working set stays L2-resident.
// ---------------------------------------------------------------------------
__global__ __launch_bounds__(256) void k_gemm_x(const unsigned short* __restrict__ Xo,
                                                const unsigned short* __restrict__ WTo,
                                                const float* __restrict__ bias,
                                                float* __restrict__ Mx) {
  const __bf16* Xb = (const __bf16*)Xo;
  const __bf16* WT = (const __bf16*)WTo;
  const int lane = threadIdx.x & 31;
  const int wave = threadIdx.x >> 5;
  const int mw = wave & 3, nw = wave >> 2;
  const int Mb = blockIdx.y * 128 + mw * 32;
  const int Nb = blockIdx.x * 128 + nw * 64;
  const int l16 = lane & 15;
  const int offA = (lane >> 4) * 16;  // A-frag K sub-offset
  const int offB = (lane >> 4) * 16;  // B-frag K sub-offset

  const __bf16* a0p = Xb + (size_t)(Mb + l16) * U_ + offA;
  const __bf16* a1p = a0p + 16 * U_;
  const __bf16* bp[4];
#pragma unroll
  for (int j = 0; j < 4; ++j)
    bp[j] = WT + (size_t)(Nb + j * 16 + l16) * TWOU_ + offB;

  v8f acc0[4] = {v8f{}, v8f{}, v8f{}, v8f{}};
  v8f acc1[4] = {v8f{}, v8f{}, v8f{}, v8f{}};
  for (int k = 0; k < U_; k += 32) {
    v16bf a0 = load_a_bf16(a0p + k);
    v16bf a1 = load_a_bf16(a1p + k);
#pragma unroll
    for (int j = 0; j < 4; ++j) {
      v16bf b = *(const v16bf*)(bp[j] + k);
      acc0[j] = wmma_bf16(a0, b, acc0[j]);
      acc1[j] = wmma_bf16(a1, b, acc1[j]);
    }
  }
#pragma unroll
  for (int j = 0; j < 4; ++j) {
    const int c = Nb + j * 16 + l16;
    const float bc = bias[c];
#pragma unroll
    for (int v = 0; v < 8; ++v) {
      // C/D layout: M = v + 8*(lane/16), N = lane%16
      int r0 = Mb + v + ((lane >> 4) * 8);
      __builtin_nontemporal_store(acc0[j][v] + bc, &Mx[(size_t)r0 * TWOU_ + c]);
      __builtin_nontemporal_store(acc1[j][v] + bc, &Mx[(size_t)(r0 + 16) * TWOU_ + c]);
    }
  }
}

// ---------------------------------------------------------------------------
// Kernel 5: one recurrence step.
//   m_g = h @ W_h[:, u],  m_c = h @ W_h[:, 1024+u]
//   h'  = sigmoid(m_g + Mx_g + 1) * h + (1 - sigmoid) * tanh(m_c + Mx_c)
// A-operand comes pre-converted to bf16 (written by the previous step's
// epilogue) -> zero conversion VALU on the serial critical path.
// One shared A-fragment feeds both accumulators; pointwise update fused
// into the epilogue, which emits h' as fp32 (blend state), bf16 (next A),
// and the NT-streamed output slice.
// ---------------------------------------------------------------------------
__global__ __launch_bounds__(256) void k_step(const unsigned short* __restrict__ WTo,
                                              const float* __restrict__ Mx,
                                              const unsigned short* __restrict__ hbin,
                                              const float* __restrict__ hfin,
                                              unsigned short* __restrict__ hbout,
                                              float* __restrict__ hfout,
                                              float* __restrict__ out,
                                              int t) {
  const __bf16* WT = (const __bf16*)WTo;
  const __bf16* hb = (const __bf16*)hbin;
  const int lane = threadIdx.x & 31;
  const int wave = threadIdx.x >> 5;
  const int mw = wave & 3;   // row tile (16 rows each)
  const int uw = wave >> 2;  // u sub-tile (16 cols each)
  const int u0 = blockIdx.x * 32 + uw * 16;
  const int l16 = lane & 15;
  const int offA = (lane >> 4) * 16;
  const int offB = (lane >> 4) * 16;
  const int cg = u0 + l16;

  const __bf16* ap = hb + (size_t)(mw * 16 + l16) * U_ + offA;
  // h contributes K-indices 1024..2047 of the concatenated input.
  const __bf16* bgp = WT + (size_t)cg * TWOU_ + U_ + offB;
  const __bf16* bcp = WT + (size_t)(U_ + cg) * TWOU_ + U_ + offB;

  v8f accg = {}, accc = {};
  for (int k = 0; k < U_; k += 32) {
    v16bf a  = load_a_bf16(ap + k);
    v16bf bg = *(const v16bf*)(bgp + k);
    v16bf bc = *(const v16bf*)(bcp + k);
    accg = wmma_bf16(a, bg, accg);
    accc = wmma_bf16(a, bc, accc);
  }
#pragma unroll
  for (int v = 0; v < 8; ++v) {
    int b = mw * 16 + v + ((lane >> 4) * 8);
    int u = u0 + l16;
    const float* mrow = Mx + ((size_t)b * T_ + t) * TWOU_;
    float mg = accg[v] + __builtin_nontemporal_load(&mrow[u]);
    float mc = accc[v] + __builtin_nontemporal_load(&mrow[U_ + u]);
    float g = 1.0f / (1.0f + __expf(-(mg + 1.0f)));
    float e = __expf(2.0f * mc);
    float c = 1.0f - 2.0f / (e + 1.0f);  // tanh(mc), saturates correctly
    float hold = hfin[(size_t)b * U_ + u];
    float hn = g * hold + (1.0f - g) * c;
    hfout[(size_t)b * U_ + u] = hn;
    ((__bf16*)hbout)[(size_t)b * U_ + u] = (__bf16)hn;
    __builtin_nontemporal_store(hn, &out[((size_t)b * T_ + t) * U_ + u]);
  }
}

// ---------------------------------------------------------------------------
extern "C" void kernel_launch(void* const* d_in, const int* in_sizes, int n_in,
                              void* d_out, int out_size, void* d_ws, size_t ws_size,
                              hipStream_t stream) {
  (void)in_sizes; (void)n_in; (void)out_size; (void)ws_size;
  const float* x    = (const float*)d_in[0];
  const float* h0   = (const float*)d_in[1];
  const float* W    = (const float*)d_in[2];
  const float* bias = (const float*)d_in[3];
  float* out = (float*)d_out;

  char* ws = (char*)d_ws;
  const size_t WT_BYTES = (size_t)TWOU_ * TWOU_ * 2;        // 8 MB
  const size_t XB_BYTES = (size_t)B_ * T_ * U_ * 2;         // 64 MB
  const size_t MX_BYTES = (size_t)B_ * T_ * TWOU_ * 4;      // 256 MB
  const size_t HF_BYTES = (size_t)B_ * U_ * 4;              // 256 KB each
  unsigned short* WT = (unsigned short*)ws;
  unsigned short* Xb = (unsigned short*)(ws + WT_BYTES);
  float* Mx  = (float*)(ws + WT_BYTES + XB_BYTES);
  float* hfA = (float*)(ws + WT_BYTES + XB_BYTES + MX_BYTES);
  float* hfB = (float*)(ws + WT_BYTES + XB_BYTES + MX_BYTES + HF_BYTES);
  unsigned short* hbA = (unsigned short*)(ws + WT_BYTES + XB_BYTES + MX_BYTES + 2 * HF_BYTES);
  unsigned short* hbB = hbA + (size_t)B_ * U_;

  k_wconv<<<dim3(TWOU_ / 32, TWOU_ / 32), dim3(32, 8), 0, stream>>>(W, WT);
  k_xconv<<<(unsigned)((size_t)B_ * T_ * U_ / 4 / 256), 256, 0, stream>>>(x, Xb);
  k_hinit<<<(B_ * U_) / 256, 256, 0, stream>>>(h0, hfA, hbA);
  k_gemm_x<<<dim3(TWOU_ / 128, (B_ * T_) / 128), 256, 0, stream>>>(Xb, WT, bias, Mx);

  for (int t = 0; t < T_; ++t) {
    const unsigned short* hbin = (t & 1) ? hbB : hbA;
    const float*          hfin = (t & 1) ? hfB : hfA;
    unsigned short* hbout = (t & 1) ? hbA : hbB;
    float*          hfout = (t & 1) ? hfA : hfB;
    k_step<<<U_ / 32, 256, 0, stream>>>(WT, Mx, hbin, hfin, hbout, hfout, out, t);
  }
}
